// Lenia_Flow_47339129536931
// MI455X (gfx1250) — matrix operand order repaired and checked by reading, not compiled, not run on hardware
//
#include <hip/hip_runtime.h>
#include <hip/hip_bf16.h>

// ---------------------------------------------------------------------------
// Flow-Lenia forward step, gfx1250 (MI455X), wave32.
//   X=Y=512, C=3, NK=15. FFT-based circular convolution (Stockham radix-2,
//   LDS resident), WMMA f16->f32 for the [XY,15]x[15,3] einsum, TDM tile
//   loads for the transposes, async-to-LDS halo loads for reintegration.
// ---------------------------------------------------------------------------

#define NPIX   262144      // 512*512
#define NK     15
#define SIGMA_ 0.95f
#define INV_AREA (1.0f / (4.0f * SIGMA_ * SIGMA_))

typedef __attribute__((ext_vector_type(16))) _Float16 v16h;
typedef __attribute__((ext_vector_type(8)))  float    v8f;
typedef __attribute__((ext_vector_type(4)))  unsigned int v4u;
typedef __attribute__((ext_vector_type(8)))  int      v8i_;
typedef __attribute__((ext_vector_type(4)))  int      v4i_;

#define HAVE_TDM (__has_builtin(__builtin_amdgcn_tensor_load_to_lds) && \
                  __has_builtin(__builtin_amdgcn_s_wait_tensorcnt))

// Async global->LDS copy (GVS mode: 64-bit SGPR base + 32-bit VGPR offset).
// vdst = LDS byte offset VGPR, tracked on ASYNCcnt.
__device__ __forceinline__ void async_ld_b32(unsigned int lds_off,
                                             unsigned long long base,
                                             unsigned int goff) {
    asm volatile("global_load_async_to_lds_b32 %0, %1, %2"
                 :: "v"(lds_off), "v"(goff), "s"(base)
                 : "memory");
}
__device__ __forceinline__ void async_wait0() {
    asm volatile("s_wait_asynccnt 0" ::: "memory");
}

// ---------------------------------------------------------------------------
// Kernel-sum reduction (deterministic two stage, no atomics)
// ---------------------------------------------------------------------------
__global__ void k_ksum_partial(const float* __restrict__ Ks, float* __restrict__ partials) {
    const int k = blockIdx.y;         // 0..14
    const int chunk = blockIdx.x;     // 0..63, 4096 pixels each
    __shared__ float red[256];
    float s = 0.f;
    for (int t = threadIdx.x; t < 4096; t += 256) {
        int xy = chunk * 4096 + t;
        s += Ks[(size_t)xy * NK + k];
    }
    red[threadIdx.x] = s;
    __syncthreads();
    for (int off = 128; off > 0; off >>= 1) {
        if (threadIdx.x < off) red[threadIdx.x] += red[threadIdx.x + off];
        __syncthreads();
    }
    if (threadIdx.x == 0) partials[k * 64 + chunk] = red[0];
}

__global__ void k_ksum_final(const float* __restrict__ partials, float* __restrict__ invsum) {
    const int k = blockIdx.x;
    __shared__ float red[64];
    red[threadIdx.x] = partials[k * 64 + threadIdx.x];
    __syncthreads();
    for (int off = 32; off > 0; off >>= 1) {
        if (threadIdx.x < off) red[threadIdx.x] += red[threadIdx.x + off];
        __syncthreads();
    }
    if (threadIdx.x == 0) invsum[k] = 1.0f / red[0];
}

// ---------------------------------------------------------------------------
// 512-point Stockham radix-2 FFT core (ping-pong in LDS, 256 threads/row).
// sign = -1 forward, +1 inverse. Returns pointer to result buffer.
// ---------------------------------------------------------------------------
__device__ __forceinline__ float2* fft512_core(float2* bufA, float2* bufB,
                                               int tid, float sign) {
    float2* Xb = bufA;
    float2* Yb = bufB;
    int s = 1;
#pragma unroll 1
    for (int st = 0; st < 9; ++st) {
        const int n = 512 >> st;
        const int m = n >> 1;
        const int p = tid >> st;        // tid / s
        const int q = tid & (s - 1);    // tid % s
        const float theta = 6.28318530717958647692f * (float)p / (float)n;
        float sn, cs;
        __sincosf(theta, &sn, &cs);
        const float wr = cs, wi = sign * sn;
        const float2 a = Xb[q + s * p];
        const float2 b = Xb[q + s * (p + m)];
        const float2 apb = make_float2(a.x + b.x, a.y + b.y);
        const float2 amb = make_float2(a.x - b.x, a.y - b.y);
        Yb[q + s * (2 * p)]     = apb;
        Yb[q + s * (2 * p + 1)] = make_float2(amb.x * wr - amb.y * wi,
                                              amb.x * wi + amb.y * wr);
        __syncthreads();
        float2* t = Xb; Xb = Yb; Yb = t;
        s <<= 1;
    }
    return Xb;   // after 9 swaps: bufB
}

// Forward row FFT with fused real load of Ksn (planes 0..14) or x (15..17).
__global__ void k_fft_fwd_load(const float* __restrict__ Ks, const float* __restrict__ x,
                               const float* __restrict__ invsum, float2* __restrict__ out) {
    __shared__ float2 A[512];
    __shared__ float2 B[512];
    const int row = blockIdx.x, plane = blockIdx.y, tid = threadIdx.x;
    for (int e = tid; e < 512; e += 256) {
        float v;
        if (plane < NK) v = Ks[((size_t)row * 512 + e) * NK + plane] * invsum[plane];
        else            v = x[((size_t)row * 512 + e) * 3 + (plane - NK)];
        A[e] = make_float2(v, 0.f);
    }
    __syncthreads();
    float2* R = fft512_core(A, B, tid, -1.f);
    float2* o = out + (size_t)plane * NPIX + (size_t)row * 512;
    for (int e = tid; e < 512; e += 256) o[e] = R[e];
}

// Complex-to-complex row FFT (blockIdx.y = plane).
__global__ void k_fft_c2c(const float2* __restrict__ in, float2* __restrict__ out,
                          float sign, float scale) {
    __shared__ float2 A[512];
    __shared__ float2 B[512];
    const int row = blockIdx.x, plane = blockIdx.y, tid = threadIdx.x;
    const float2* ip = in + (size_t)plane * NPIX + (size_t)row * 512;
    for (int e = tid; e < 512; e += 256) A[e] = ip[e];
    __syncthreads();
    float2* R = fft512_core(A, B, tid, sign);
    float2* o = out + (size_t)plane * NPIX + (size_t)row * 512;
    for (int e = tid; e < 512; e += 256)
        o[e] = make_float2(R[e].x * scale, R[e].y * scale);
}

// Final inverse row FFT: take real part, apply 1/512 and the fftshift roll
// (+256 circular in both dims, from the shift theorem).
__global__ void k_fft_inv_c2r_shift(const float2* __restrict__ in, float* __restrict__ U) {
    __shared__ float2 A[512];
    __shared__ float2 B[512];
    const int row = blockIdx.x, plane = blockIdx.y, tid = threadIdx.x;
    const float2* ip = in + (size_t)plane * NPIX + (size_t)row * 512;
    for (int e = tid; e < 512; e += 256) A[e] = ip[e];
    __syncthreads();
    float2* R = fft512_core(A, B, tid, 1.f);
    const int xs = (row + 256) & 511;
    float* o = U + (size_t)plane * NPIX + (size_t)xs * 512;
    for (int e = tid; e < 512; e += 256)
        o[(e + 256) & 511] = R[e].x * (1.0f / 512.0f);
}

// ---------------------------------------------------------------------------
// 512x512 complex transpose, 32x32 LDS tiles (blockIdx.z = plane).
// Tile load done by the Tensor Data Mover: one wave issues a 2D D# descriptor
// (data_size=8B, tile 32x32, tensor stride 512, LDS row padding 8B per 256B
// to reproduce the [32][33] bank-conflict padding), then s_wait_tensorcnt.
// ---------------------------------------------------------------------------
__global__ void k_transpose(const float2* __restrict__ in, float2* __restrict__ out) {
    __shared__ float2 tile[32][33];
    const size_t base = (size_t)blockIdx.z * NPIX;
    const int x0 = blockIdx.x * 32, y0 = blockIdx.y * 32;
#if HAVE_TDM
    if (threadIdx.y == 0) {   // wave 0 of the block issues the TDM descriptor
        const unsigned long long gaddr =
            (unsigned long long)(const void*)(in + base + (size_t)x0 * 512 + y0);
        const unsigned int ldsoff = (unsigned int)(size_t)(void*)&tile[0][0];
        // D# group 0: count=1, lds_addr, global_addr, type=2 ("image")
        v4u g0 = { 1u,
                   ldsoff,
                   (unsigned int)gaddr,
                   (unsigned int)((gaddr >> 32) & 0x1FFFFFFu) | (2u << 30) };
        // D# group 1: data_size=3 (8B), pad_enable, pad_interval=64 DW,
        // pad_amount=2 DW; tensor_dim0=512, tensor_dim1=512; tile 32x32;
        // dim0_stride=512; dim1_stride=262144.
        v8i_ g1 = { (int)((3u << 16) | (1u << 20) | (5u << 22) | (1u << 25)),
                    (int)(512u << 16),   // tensor_dim0[15:0] at bits[63:48]
                    (int)(512u << 16),   // dim0 hi=0, tensor_dim1[15:0] at [95:80]
                    (int)(32u << 16),    // dim1 hi=0, tile_dim0=32
                    (int)(32u),          // tile_dim1=32, tile_dim2=0
                    (int)(512),          // tensor_dim0_stride[31:0]
                    0,                   // stride0 hi, dim1_stride[15:0]=0
                    4 };                 // dim1_stride[47:16] = 262144>>16
        v4i_ g2 = { 0, 0, 0, 0 };
        v4i_ g3 = { 0, 0, 0, 0 };
        v8i_ g4 = { 0, 0, 0, 0, 0, 0, 0, 0 };   // extra group (6-arg form), zero-filled
        __builtin_amdgcn_tensor_load_to_lds(g0, g1, g2, g3, g4, 0);
        __builtin_amdgcn_s_wait_tensorcnt(0);
    }
    __syncthreads();
#else
    for (int r = threadIdx.y; r < 32; r += 8)
        tile[r][threadIdx.x] = in[base + (size_t)(x0 + r) * 512 + (y0 + threadIdx.x)];
    __syncthreads();
#endif
    for (int r = threadIdx.y; r < 32; r += 8)
        out[base + (size_t)(y0 + r) * 512 + (x0 + threadIdx.x)] = tile[threadIdx.x][r];
}

// Pointwise spectral product: P[k] = FK[k] * FX[c0[k]].
__global__ void k_specmul(const float2* __restrict__ FREQ, const int* __restrict__ c0,
                          float2* __restrict__ P) {
    const int k = blockIdx.y;
    const int idx = blockIdx.x * 256 + threadIdx.x;
    const float2 a = FREQ[(size_t)k * NPIX + idx];
    const float2 b = FREQ[(size_t)(NK + c0[k]) * NPIX + idx];
    P[(size_t)k * NPIX + idx] = make_float2(a.x * b.x - a.y * b.y,
                                            a.x * b.y + a.y * b.x);
}

// ---------------------------------------------------------------------------
// Growth: Gs = (exp(-0.5*((U-m)/s)^2)*2 - 1)*h, written as f16 in the packed
// per-pixel [16] layout the WMMA A operand expects (k=15 slot zero).
// ---------------------------------------------------------------------------
__global__ void k_growth(const float* __restrict__ U, const float* __restrict__ m,
                         const float* __restrict__ s, const float* __restrict__ h,
                         _Float16* __restrict__ Gsh) {
    const int xy = blockIdx.x * 256 + threadIdx.x;
    union { _Float16 hh[16]; uint4 u4[2]; } pk;
#pragma unroll
    for (int k = 0; k < NK; ++k) {
        const float u = U[(size_t)k * NPIX + xy];
        const float t = (u - m[k]) / s[k];
        pk.hh[k] = (_Float16)((__expf(-0.5f * t * t) * 2.0f - 1.0f) * h[k]);
    }
    pk.hh[15] = (_Float16)0.f;
    uint4* dst = reinterpret_cast<uint4*>(Gsh + (size_t)xy * 16);
    dst[0] = pk.u4[0];
    dst[1] = pk.u4[1];
}

// ---------------------------------------------------------------------------
// Einsum Hs[xy,c] = sum_k Gs[xy,k] * w_c1[k,c] via v_wmma_f32_16x16x32_f16.
// One wave processes 4 consecutive 16-pixel tiles (B operand built once and
// reused across the 4 WMMAs). K padded 15->32, N padded 3->16.
// A layout (16-bit A 16x32): lanes 0-15 hold M=lane, K={0..7}; lanes 16-31
// hold M=lane-16, K={8..15}; upper K half (16..31) is zero.
// ---------------------------------------------------------------------------
__global__ void k_einsum_wmma(const _Float16* __restrict__ Gsh,
                              const float* __restrict__ w_c1,
                              float* __restrict__ Hs) {
    const int lane  = threadIdx.x & 31;
    const int wave  = threadIdx.x >> 5;
    const int tile0 = (blockIdx.x * 8 + wave) * 4;
    const int mrow  = lane & 15;
    const int hi    = lane >> 4;                   // 0: K 0..7, 1: K 8..15

    // B operand (32x16): column N=lane%16; element e<8 holds K=8*hi+e,
    // elements 8..15 hold K=16+8*hi+(e-8) (all zero here since K>=15 is padding).
    v16h b;
#pragma unroll
    for (int e = 0; e < 16; ++e) b[e] = (_Float16)0.f;
    if (mrow < 3) {
#pragma unroll
        for (int e = 0; e < 8; ++e) { const int kk = hi * 8 + e; if (kk < NK) b[e] = (_Float16)w_c1[kk * 3 + mrow]; }
    }

#pragma unroll
    for (int t = 0; t < 4; ++t) {
        const int tile = tile0 + t;
        // A operand: 8 contiguous halves of this pixel's packed Gs row.
        union { uint4 u4; _Float16 hh[8]; } ld;
        ld.u4 = *reinterpret_cast<const uint4*>(
                    Gsh + ((size_t)tile * 16 + mrow) * 16 + hi * 8);
        v16h a;
#pragma unroll
        for (int e = 0; e < 8; ++e) a[e] = ld.hh[e];
#pragma unroll
        for (int e = 8; e < 16; ++e) a[e] = (_Float16)0.f;

        v8f c = {};
        c = __builtin_amdgcn_wmma_f32_16x16x32_f16(
                /*neg_a=*/false, a, /*neg_b=*/false, b,
                /*c_mod=*/(short)0, c, /*reuse_a=*/false, /*reuse_b=*/false);

        // D layout: lane<16 -> M=r, N=lane; lane>=16 -> M=8+r, N=lane-16.
        if (mrow < 3) {
            const int Mb = hi * 8;
#pragma unroll
            for (int r = 0; r < 8; ++r)
                Hs[((size_t)tile * 16 + Mb + r) * 3 + mrow] = c[r];
        }
    }
}

// ---------------------------------------------------------------------------
// Sobel(Hs), Sobel(sum_c x), alpha, flow F, and mu = pos + clip(DT*F).
// Zero-padded 'SAME' stencils (matches torch conv2d padding).
// ---------------------------------------------------------------------------
__global__ void k_flow(const float* __restrict__ x, const float* __restrict__ Hs,
                       float* __restrict__ MU) {
    const int idx = blockIdx.x * 256 + threadIdx.x;
    const int i = idx >> 9, j = idx & 511;
    float g0H[3] = {0.f, 0.f, 0.f};   // gradient along axis 0 (i) of Hs
    float g1H[3] = {0.f, 0.f, 0.f};   // gradient along axis 1 (j) of Hs
    float g0X = 0.f, g1X = 0.f;       // gradients of sum_c x
#pragma unroll
    for (int di = -1; di <= 1; ++di) {
#pragma unroll
        for (int dj = -1; dj <= 1; ++dj) {
            const int ii = i + di, jj = j + dj;
            if (ii < 0 || ii > 511 || jj < 0 || jj > 511) continue;
            const float w0 = (float)di * ((dj == 0) ? 2.f : 1.f);  // kx^T
            const float w1 = (float)dj * ((di == 0) ? 2.f : 1.f);  // kx
            const size_t nb = ((size_t)ii * 512 + jj) * 3;
            float xs = 0.f;
#pragma unroll
            for (int c = 0; c < 3; ++c) {
                const float hv = Hs[nb + c];
                g0H[c] += w0 * hv;
                g1H[c] += w1 * hv;
                xs += x[nb + c];
            }
            g0X += w0 * xs;
            g1X += w1 * xs;
        }
    }
    const float ma = 2.0f - SIGMA_;          // 1.05
    const float p0 = (float)i + 0.5f, p1 = (float)j + 0.5f;
#pragma unroll
    for (int c = 0; c < 3; ++c) {
        const float xv = x[(size_t)idx * 3 + c];
        float al = xv * 0.5f;                 // (x/THETA_X)^N_EXP, THETA=2, N=2
        al = al * al;
        al = fminf(al, 1.0f);
        const float F0 = g0H[c] * (1.f - al) - g0X * al;
        const float F1 = g1H[c] * (1.f - al) - g1X * al;
        float mu0 = p0 + fminf(fmaxf(0.2f * F0, -ma), ma);
        float mu1 = p1 + fminf(fmaxf(0.2f * F1, -ma), ma);
        mu0 = fminf(fmaxf(mu0, SIGMA_), 512.f - SIGMA_);
        mu1 = fminf(fmaxf(mu1, SIGMA_), 512.f - SIGMA_);
        MU[((size_t)idx * 2 + 0) * 3 + c] = mu0;
        MU[((size_t)idx * 2 + 1) * 3 + c] = mu1;
    }
}

// ---------------------------------------------------------------------------
// Reintegration tracking: 25 circular offsets, LDS halo tile (20x20 cells,
// 9 floats per cell: grid[3] + mu[2][3]). Halo loaded with async-to-LDS
// (GLOBAL_LOAD_ASYNC_TO_LDS_B32, tracked on ASYNCcnt) instead of a VGPR
// round trip.
// ---------------------------------------------------------------------------
__global__ void k_reintegrate(const float* __restrict__ x, const float* __restrict__ MU,
                              float* __restrict__ out) {
    __shared__ float gt[20][20][3];
    __shared__ float mt[20][20][6];
    const int bx = blockIdx.x * 16, by = blockIdx.y * 16;
    const int tid = threadIdx.y * 16 + threadIdx.x;

    const unsigned long long xbase  = (unsigned long long)(const void*)x;
    const unsigned long long mubase = (unsigned long long)(const void*)MU;
    const unsigned int gt0 = (unsigned int)(size_t)(void*)&gt[0][0][0];
    const unsigned int mt0 = (unsigned int)(size_t)(void*)&mt[0][0][0];

    for (int t = tid; t < 400; t += 256) {
        const int u = t / 20, v = t % 20;
        const int gi = (bx + u - 2) & 511;
        const int gj = (by + v - 2) & 511;
        const unsigned int g = (unsigned int)(gi * 512 + gj);
#pragma unroll
        for (int c = 0; c < 3; ++c)
            async_ld_b32(gt0 + ((unsigned int)t * 3u + c) * 4u, xbase,  (g * 3u + c) * 4u);
#pragma unroll
        for (int e = 0; e < 6; ++e)
            async_ld_b32(mt0 + ((unsigned int)t * 6u + e) * 4u, mubase, (g * 6u + e) * 4u);
    }
    async_wait0();       // each wave drains its own ASYNCcnt
    __syncthreads();     // then all waves see the full LDS tile

    const int i = bx + threadIdx.y, j = by + threadIdx.x;
    const float p0 = (float)i + 0.5f, p1 = (float)j + 0.5f;
    float acc0 = 0.f, acc1 = 0.f, acc2 = 0.f;
#pragma unroll
    for (int du = 0; du < 5; ++du) {
#pragma unroll
        for (int dv = 0; dv < 5; ++dv) {
            const int u = threadIdx.y + du, v = threadIdx.x + dv;
#pragma unroll
            for (int c = 0; c < 3; ++c) {
                float s0 = 0.5f - fabsf(p0 - mt[u][v][c])     + SIGMA_;
                float s1 = 0.5f - fabsf(p1 - mt[u][v][3 + c]) + SIGMA_;
                s0 = fminf(fmaxf(s0, 0.f), 1.f);
                s1 = fminf(fmaxf(s1, 0.f), 1.f);
                const float a = gt[u][v][c] * (s0 * s1 * INV_AREA);
                if (c == 0) acc0 += a; else if (c == 1) acc1 += a; else acc2 += a;
            }
        }
    }
    const size_t o = ((size_t)i * 512 + j) * 3;
    out[o + 0] = acc0;
    out[o + 1] = acc1;
    out[o + 2] = acc2;
}

// ---------------------------------------------------------------------------
// Launch
// ---------------------------------------------------------------------------
extern "C" void kernel_launch(void* const* d_in, const int* in_sizes, int n_in,
                              void* d_out, int out_size, void* d_ws, size_t ws_size,
                              hipStream_t stream) {
    const float* x    = (const float*)d_in[0];   // [512,512,3]
    const float* Ks   = (const float*)d_in[1];   // [512,512,15]
    const float* m    = (const float*)d_in[2];   // [15]
    const float* s    = (const float*)d_in[3];   // [15]
    const float* h    = (const float*)d_in[4];   // [15]
    const float* w_c1 = (const float*)d_in[5];   // [15,3]
    const int*   c0   = (const int*)  d_in[6];   // [15]
    float* out = (float*)d_out;

    char* wsb = (char*)d_ws;
    float*  partials = (float*)wsb;                              //  3840 B
    float*  invsum   = (float*)(wsb + 4096);                     //    60 B
    float2* B1       = (float2*)(wsb + 8192);                    // 18 planes c64
    float2* B2       = B1 + (size_t)18 * NPIX;                   // 18 planes c64
    float*  U        = (float*)(B2 + (size_t)18 * NPIX);         // 15 planes f32
    // Overlays on B1 (free after the last transpose consumed it):
    _Float16* Gsh = (_Float16*)B1;                               // [NPIX,16] f16
    float*    Hs  = (float*)(Gsh + (size_t)16 * NPIX);           // [NPIX,3]
    float*    MU  = Hs + (size_t)3 * NPIX;                       // [NPIX,2,3]

    // 1-2: kernel sums -> 1/sum (deterministic)
    k_ksum_partial<<<dim3(64, NK), 256, 0, stream>>>(Ks, partials);
    k_ksum_final<<<NK, 64, 0, stream>>>(partials, invsum);
    // 3: forward row FFT (along y) with fused Ksn / x load   -> B1 [i][v]
    k_fft_fwd_load<<<dim3(512, 18), 256, 0, stream>>>(Ks, x, invsum, B1);
    // 4: transpose (TDM tile loads)                           -> B2 [v][i]
    k_transpose<<<dim3(16, 16, 18), dim3(32, 8), 0, stream>>>(B1, B2);
    // 5: forward row FFT (along x)                            -> B1 = FREQ [v][u]
    k_fft_c2c<<<dim3(512, 18), 256, 0, stream>>>(B2, B1, -1.f, 1.f);
    // 6: spectral product FK[k]*FX[c0[k]]                     -> B2 planes 0..14
    k_specmul<<<dim3(1024, NK), 256, 0, stream>>>(B1, c0, B2);
    // 7: inverse row FFT (along u), scale 1/512               -> B1 [v][x]
    k_fft_c2c<<<dim3(512, NK), 256, 0, stream>>>(B2, B1, 1.f, 1.f / 512.f);
    // 8: transpose (TDM tile loads)                           -> B2 [x][v]
    k_transpose<<<dim3(16, 16, NK), dim3(32, 8), 0, stream>>>(B1, B2);
    // 9: inverse row FFT (along v), real part + fftshift roll -> U [k][x][y]
    k_fft_inv_c2r_shift<<<dim3(512, NK), 256, 0, stream>>>(B2, U);
    // 10: growth -> packed f16 Gs (WMMA A layout)
    k_growth<<<1024, 256, 0, stream>>>(U, m, s, h, Gsh);
    // 11: WMMA einsum -> Hs (4 tiles per wave)
    k_einsum_wmma<<<512, 256, 0, stream>>>(Gsh, w_c1, Hs);
    // 12: sobel + alpha + flow -> mu
    k_flow<<<1024, 256, 0, stream>>>(x, Hs, MU);
    // 13: reintegration -> out
    k_reintegrate<<<dim3(32, 32), dim3(16, 16), 0, stream>>>(x, MU, out);
}
